// MistralQuantizedFlashAttention2_43035572306189
// MI455X (gfx1250) — matrix operand order repaired and checked by reading, not compiled, not run on hardware
//
#include <hip/hip_runtime.h>

// ---------------------------------------------------------------------------
// Types
// ---------------------------------------------------------------------------
typedef __bf16 bf16;
typedef __attribute__((ext_vector_type(8)))  __bf16 bf16x8;
typedef __attribute__((ext_vector_type(16))) __bf16 bf16x16;
typedef __attribute__((ext_vector_type(8)))  float  f32x8;
typedef __attribute__((ext_vector_type(4)))  unsigned int u32x4;
typedef __attribute__((ext_vector_type(8)))  int i32x8;
typedef __attribute__((ext_vector_type(4)))  int i32x4;

#define HIDDEN   4096
#define SEQ      2048
#define NHEADS   32
#define NKV      8
#define HDIM     128
#define ATT_SCALE 0.08838834764831845f   // 1/sqrt(128)

__device__ __forceinline__ bf16x8 ld8(const bf16* p) {
    return *(const bf16x8*)p;
}
__device__ __forceinline__ bf16x16 cat8(bf16x8 a, bf16x8 b) {
    return __builtin_shufflevector(a, b, 0,1,2,3,4,5,6,7,8,9,10,11,12,13,14,15);
}
__device__ __forceinline__ f32x8 zero8() {
    f32x8 z = {0.f,0.f,0.f,0.f,0.f,0.f,0.f,0.f};
    return z;
}
__device__ __forceinline__ f32x8 wmma_bf16(bf16x16 a, bf16x16 b, f32x8 c) {
    return __builtin_amdgcn_wmma_f32_16x16x32_bf16(
        false, a, false, b, (short)0, c, false, false);
}

// ---------------------------------------------------------------------------
// Tensor Data Mover: 2-D bf16 tile load Global -> LDS.
// D# bitfields per CDNA5 ISA:
//   group0: count=1 | lds_addr[63:32] | global_addr[120:64] | type=2 [127:126]
//   group1: data_size=1 (2B) | tensor_dim0[79:48] | tensor_dim1[111:80]
//           | tile_dim0[127:112] | tile_dim1[143:128] | dim0_stride[207:160]
// 6-arg builtin form (clang-23 / therock-10.0): (g0, g1, g2, g3, g4, cpol)
// ---------------------------------------------------------------------------
__device__ __forceinline__ void tdm_load_2d(unsigned lds_byte_addr,
                                            const bf16* gptr,
                                            unsigned tensor_d0, unsigned tensor_d1,
                                            unsigned tile_d0, unsigned tile_d1,
                                            unsigned stride_d0) {
    unsigned long long ga = (unsigned long long)(const void*)gptr;
    u32x4 g0;
    g0.x = 1u;                                               // count=1, user D#
    g0.y = lds_byte_addr;                                    // lds_addr
    g0.z = (unsigned)ga;                                     // global_addr[31:0]
    g0.w = (unsigned)((ga >> 32) & 0x01FFFFFFu) | (2u << 30);// addr[56:32]|type=2
    i32x8 g1 = {
        (int)(1u << 16),                                     // data_size=1 (2B)
        (int)(tensor_d0 << 16),                              // tensor_dim0 lo16
        (int)((tensor_d0 >> 16) | (tensor_d1 << 16)),        // d0 hi | d1 lo
        (int)((tensor_d1 >> 16) | (tile_d0 << 16)),          // d1 hi | tile0
        (int)(tile_d1 & 0xFFFFu),                            // tile1, tile2=0
        (int)stride_d0,                                      // dim0_stride lo32
        0, 0
    };
    i32x4 z4 = {0, 0, 0, 0};
    i32x8 z8 = {0, 0, 0, 0, 0, 0, 0, 0};
    __builtin_amdgcn_tensor_load_to_lds(g0, g1, z4, z4, z8, 0);
}
__device__ __forceinline__ unsigned lds_off(const void* p) {
    return (unsigned)(size_t)p;     // LDS aperture: addr[31:0] is the LDS offset
}

// ---------------------------------------------------------------------------
// 1) fp32 -> bf16 conversion
// ---------------------------------------------------------------------------
__global__ void f32_to_bf16_kernel(const float* __restrict__ in,
                                   bf16* __restrict__ out, int n) {
    int i = blockIdx.x * blockDim.x + threadIdx.x;
    if (i < n) out[i] = (bf16)in[i];
}

// ---------------------------------------------------------------------------
// 2) GEMM: C[M,N](f32) = A[M,K](bf16 row-major) * B[N,K](bf16 row-major)^T
//    Block 128 thr = 4 waves; block tile 64x64; wave tile 16x64.
//    K-loop step 32, software-pipelined: next iteration's fragments are
//    loaded before the current WMMAs so waits are partial (no full drains).
// ---------------------------------------------------------------------------
__global__ __launch_bounds__(128)
void gemm_bf16_kernel(const bf16* __restrict__ A, const bf16* __restrict__ B,
                      float* __restrict__ C, int K, int ldc) {
    const int lane = threadIdx.x & 31;
    const int wave = threadIdx.x >> 5;
    const int lrow = lane & 15;
    const int lhi  = lane >> 4;
    const int mrow  = blockIdx.y * 64 + wave * 16;
    const int nbase = blockIdx.x * 64;

    f32x8 acc[4];
    #pragma unroll
    for (int t = 0; t < 4; ++t) acc[t] = zero8();

    const bf16* arow  = A + (size_t)(mrow + lrow) * K + (lhi ? 16 : 0);
    const bf16* brow0 = B + (size_t)(nbase + lrow) * K + lhi * 16;

    // prologue: fragments for k = 0
    bf16x8 a0 = ld8(arow), a1 = ld8(arow + 8);
    bf16x8 b0[4], b1[4];
    #pragma unroll
    for (int t = 0; t < 4; ++t) {
        const bf16* bp = brow0 + (size_t)t * 16 * K;
        b0[t] = ld8(bp); b1[t] = ld8(bp + 8);
    }
    for (int k = 32; k < K; k += 32) {
        // issue next-k loads first (overlap with current WMMAs)
        bf16x8 na0 = ld8(arow + k), na1 = ld8(arow + k + 8);
        bf16x8 nb0[4], nb1[4];
        #pragma unroll
        for (int t = 0; t < 4; ++t) {
            const bf16* bp = brow0 + (size_t)t * 16 * K + k;
            nb0[t] = ld8(bp); nb1[t] = ld8(bp + 8);
        }
        __builtin_prefetch(arow + k + 256, 0, 1);
        bf16x16 af = cat8(a0, a1);
        #pragma unroll
        for (int t = 0; t < 4; ++t)
            acc[t] = wmma_bf16(af, cat8(b0[t], b1[t]), acc[t]);
        a0 = na0; a1 = na1;
        #pragma unroll
        for (int t = 0; t < 4; ++t) { b0[t] = nb0[t]; b1[t] = nb1[t]; }
    }
    {   // epilogue: last k-step
        bf16x16 af = cat8(a0, a1);
        #pragma unroll
        for (int t = 0; t < 4; ++t)
            acc[t] = wmma_bf16(af, cat8(b0[t], b1[t]), acc[t]);
    }
    #pragma unroll
    for (int t = 0; t < 4; ++t)
        #pragma unroll
        for (int r = 0; r < 8; ++r)
            C[(size_t)(mrow + r + lhi * 8) * ldc + nbase + t * 16 + lrow] = acc[t][r];
}

// ---------------------------------------------------------------------------
// 3) RoPE + pack:  Qf->Qb[32][2048][128], Kf->Kb[8][2048][128],
//                  Vf->VTb[8][128][2048] (transposed)
// ---------------------------------------------------------------------------
__global__ void rope_pack_kernel(const float* __restrict__ Qf,
                                 const float* __restrict__ Kf,
                                 const float* __restrict__ Vf,
                                 const int* __restrict__ pos,
                                 bf16* __restrict__ Qb,
                                 bf16* __restrict__ Kb,
                                 bf16* __restrict__ VTb) {
    int idx = blockIdx.x * blockDim.x + threadIdx.x;
    if (idx >= SEQ * HIDDEN) return;
    const int s   = idx >> 12;
    const int col = idx & 4095;
    const int h   = col >> 7;
    const int d   = col & 127;

    const float t  = (float)pos[s];
    const int   fi = d & 63;
    const float inv = __powf(10000.0f, -(2.0f * (float)fi) / 128.0f);
    float sv, cv;
    __sincosf(t * inv, &sv, &cv);

    {
        float x = Qf[idx];
        float o = (d < 64) ? -Qf[idx + 64] : Qf[idx - 64];
        Qb[((size_t)h * SEQ + s) * HDIM + d] = (bf16)(x * cv + o * sv);
    }
    if (col < NKV * HDIM) {
        const int kidx = s * (NKV * HDIM) + col;
        float x = Kf[kidx];
        float o = (d < 64) ? -Kf[kidx + 64] : Kf[kidx - 64];
        Kb[((size_t)h * SEQ + s) * HDIM + d] = (bf16)(x * cv + o * sv);
        VTb[((size_t)h * HDIM + d) * SEQ + s] = (bf16)Vf[kidx];
    }
}

// ---------------------------------------------------------------------------
// 4) Flash attention. Grid (32 heads, 32); block = 4 waves sharing a 64-row
//    query stripe of one head. K/V 32-key tiles are TDM-DMA'd into double-
//    buffered LDS by wave 0 (tensor_load_to_lds + s_wait_tensorcnt), shared
//    by all 4 waves; per-wave online softmax; P transposed C->A layout via a
//    small per-wave LDS tile (same-wave DS is in-order).
// ---------------------------------------------------------------------------
__global__ __launch_bounds__(128)
void flash_attn_kernel(const bf16* __restrict__ Qb, const bf16* __restrict__ Kb,
                       const bf16* __restrict__ VTb, bf16* __restrict__ attn) {
    const int h    = blockIdx.x;
    const int kvh  = h >> 2;
    const int wave = threadIdx.x >> 5;
    const int lane = threadIdx.x & 31;
    const int lrow = lane & 15;
    const int lhi  = lane >> 4;
    const int qblock = blockIdx.y * 64;
    const int qbase  = qblock + wave * 16;

    __shared__ bf16 Kt[2][32][HDIM] __attribute__((aligned(16)));   // 16 KB
    __shared__ bf16 Vt[2][HDIM][32] __attribute__((aligned(16)));   // 16 KB
    __shared__ bf16 PtAll[4][16][40] __attribute__((aligned(16)));  // 5 KB
    bf16 (*Pt)[40] = PtAll[wave];

    const bf16* Qh  = Qb  + (size_t)h   * SEQ  * HDIM;
    const bf16* Kh  = Kb  + (size_t)kvh * SEQ  * HDIM;
    const bf16* VTh = VTb + (size_t)kvh * HDIM * SEQ;

    // Q fragments for this wave's 16-row tile (4 K-chunks of 32)
    bf16x16 qfrag[4];
    {
        const bf16* qrow = Qh + (size_t)(qbase + lrow) * HDIM + (lhi ? 16 : 0);
        #pragma unroll
        for (int c = 0; c < 4; ++c)
            qfrag[c] = cat8(ld8(qrow + c * 32), ld8(qrow + c * 32 + 8));
    }

    f32x8 O[8];
    #pragma unroll
    for (int c = 0; c < 8; ++c) O[c] = zero8();
    float m[8], l[8];
    #pragma unroll
    for (int r = 0; r < 8; ++r) { m[r] = -1e30f; l[r] = 0.f; }

    const int nkb = (qblock + 95) >> 5;      // 32-key blocks (uniform per block)

    // prime buffer 0 via TDM (wave 0 only; TDM ignores EXEC, per-wave issue)
    if (wave == 0) {
        tdm_load_2d(lds_off(&Kt[0][0][0]), Kh, HDIM, SEQ, HDIM, 32, HDIM);
        tdm_load_2d(lds_off(&Vt[0][0][0]), VTh, SEQ, HDIM, 32, HDIM, SEQ);
        __builtin_amdgcn_s_wait_tensorcnt(0);
    }
    __syncthreads();

    int buf = 0;
    for (int kb = 0; kb < nkb; ++kb) {
        const int kbase = kb * 32;
        // kick off next tile's DMA into the other buffer
        const bool have_next = (kb + 1 < nkb);
        if (have_next && wave == 0) {
            const int nb = kbase + 32;
            tdm_load_2d(lds_off(&Kt[buf ^ 1][0][0]), Kh + (size_t)nb * HDIM,
                        HDIM, SEQ, HDIM, 32, HDIM);
            tdm_load_2d(lds_off(&Vt[buf ^ 1][0][0]), VTh + nb,
                        SEQ, HDIM, 32, HDIM, SEQ);
        }
        // ---- scores from LDS K tile: S0 keys +0..15, S1 keys +16..31
        f32x8 S0 = zero8(), S1 = zero8();
        const bf16* kt = &Kt[buf][0][0];
        #pragma unroll
        for (int c = 0; c < 4; ++c) {
            const bf16* kp0 = kt + lrow * HDIM + c * 32 + lhi * 16;
            S0 = wmma_bf16(qfrag[c], cat8(ld8(kp0), ld8(kp0 + 8)), S0);
            const bf16* kp1 = kt + (16 + lrow) * HDIM + c * 32 + lhi * 16;
            S1 = wmma_bf16(qfrag[c], cat8(ld8(kp1), ld8(kp1 + 8)), S1);
        }
        // ---- causal mask + online softmax (16-lane xor butterflies)
        float alpha[8];
        #pragma unroll
        for (int r = 0; r < 8; ++r) {
            const int qrow = qbase + r + lhi * 8;
            const int k0 = kbase + lrow, k1 = kbase + 16 + lrow;
            float s0 = S0[r] * ATT_SCALE + (k0 > qrow ? -1e9f : 0.f);
            float s1 = S1[r] * ATT_SCALE + (k1 > qrow ? -1e9f : 0.f);
            float pm = fmaxf(s0, s1);
            pm = fmaxf(pm, __shfl_xor(pm, 1));
            pm = fmaxf(pm, __shfl_xor(pm, 2));
            pm = fmaxf(pm, __shfl_xor(pm, 4));
            pm = fmaxf(pm, __shfl_xor(pm, 8));
            const float mn = fmaxf(m[r], pm);
            alpha[r] = __expf(m[r] - mn);
            m[r] = mn;
            const float p0 = __expf(s0 - mn);
            const float p1 = __expf(s1 - mn);
            float ps = p0 + p1;
            ps += __shfl_xor(ps, 1);
            ps += __shfl_xor(ps, 2);
            ps += __shfl_xor(ps, 4);
            ps += __shfl_xor(ps, 8);
            l[r] = l[r] * alpha[r] + ps;
            Pt[r + lhi * 8][lrow]      = (bf16)p0;
            Pt[r + lhi * 8][16 + lrow] = (bf16)p1;
        }
        #pragma unroll
        for (int c = 0; c < 8; ++c)
            #pragma unroll
            for (int r = 0; r < 8; ++r)
                O[c][r] *= alpha[r];
        // ---- P back as 16x32 A-fragment (transpose via per-wave LDS)
        bf16x16 pfrag;
        {
            const bf16* p = &Pt[lrow][lhi ? 16 : 0];
            pfrag = cat8(ld8(p), ld8(p + 8));
        }
        // ---- O += P * V^T from LDS V tile
        #pragma unroll
        for (int c = 0; c < 8; ++c) {
            const bf16* vp = &Vt[buf][c * 16 + lrow][lhi * 16];
            O[c] = wmma_bf16(pfrag, cat8(ld8(vp), ld8(vp + 8)), O[c]);
        }
        // ---- retire next tile's DMA, publish, flip buffers
        if (have_next && wave == 0)
            __builtin_amdgcn_s_wait_tensorcnt(0);
        __syncthreads();
        buf ^= 1;
    }
    // ---- normalize + store bf16 attn output [2048][4096]
    #pragma unroll
    for (int c = 0; c < 8; ++c)
        #pragma unroll
        for (int r = 0; r < 8; ++r) {
            const int row = qbase + r + lhi * 8;
            attn[(size_t)row * HIDDEN + h * HDIM + c * 16 + lrow] =
                (bf16)(O[c][r] / l[r]);
        }
}

// ---------------------------------------------------------------------------
// Host orchestration
// ---------------------------------------------------------------------------
extern "C" void kernel_launch(void* const* d_in, const int* in_sizes, int n_in,
                              void* d_out, int out_size, void* d_ws, size_t ws_size,
                              hipStream_t stream) {
    const float* X   = (const float*)d_in[0];
    const int*   pos = (const int*)d_in[2];
    const float* Wq  = (const float*)d_in[3];
    const float* Wk  = (const float*)d_in[4];
    const float* Wv  = (const float*)d_in[5];
    const float* Wo  = (const float*)d_in[6];
    float* out = (float*)d_out;

    char* ws = (char*)d_ws;
    size_t off = 0;
    auto carve = [&](size_t bytes) -> char* {
        char* p = ws + off;
        off += (bytes + 255) & ~(size_t)255;
        return p;
    };
    bf16* Xb    = (bf16*)carve((size_t)SEQ * HIDDEN * 2);
    bf16* Wqb   = (bf16*)carve((size_t)HIDDEN * HIDDEN * 2);
    bf16* Wkb   = (bf16*)carve((size_t)NKV * HDIM * HIDDEN * 2);
    bf16* Wvb   = (bf16*)carve((size_t)NKV * HDIM * HIDDEN * 2);
    bf16* Wob   = (bf16*)carve((size_t)HIDDEN * HIDDEN * 2);
    float* Qf   = (float*)carve((size_t)SEQ * HIDDEN * 4);
    float* Kf   = (float*)carve((size_t)SEQ * NKV * HDIM * 4);
    float* Vf   = (float*)carve((size_t)SEQ * NKV * HDIM * 4);
    bf16* Qb    = (bf16*)carve((size_t)NHEADS * SEQ * HDIM * 2);
    bf16* Kb    = (bf16*)carve((size_t)NKV * SEQ * HDIM * 2);
    bf16* VTb   = (bf16*)carve((size_t)NKV * HDIM * SEQ * 2);
    bf16* attnb = (bf16*)carve((size_t)SEQ * HIDDEN * 2);

    {
        int n = SEQ * HIDDEN;
        f32_to_bf16_kernel<<<(n + 255) / 256, 256, 0, stream>>>(X, Xb, n);
        n = HIDDEN * HIDDEN;
        f32_to_bf16_kernel<<<(n + 255) / 256, 256, 0, stream>>>(Wq, Wqb, n);
        f32_to_bf16_kernel<<<(n + 255) / 256, 256, 0, stream>>>(Wo, Wob, n);
        n = NKV * HDIM * HIDDEN;
        f32_to_bf16_kernel<<<(n + 255) / 256, 256, 0, stream>>>(Wk, Wkb, n);
        f32_to_bf16_kernel<<<(n + 255) / 256, 256, 0, stream>>>(Wv, Wvb, n);
    }
    gemm_bf16_kernel<<<dim3(HIDDEN / 64, SEQ / 64), 128, 0, stream>>>(
        Xb, Wqb, Qf, HIDDEN, HIDDEN);
    gemm_bf16_kernel<<<dim3((NKV * HDIM) / 64, SEQ / 64), 128, 0, stream>>>(
        Xb, Wkb, Kf, HIDDEN, NKV * HDIM);
    gemm_bf16_kernel<<<dim3((NKV * HDIM) / 64, SEQ / 64), 128, 0, stream>>>(
        Xb, Wvb, Vf, HIDDEN, NKV * HDIM);
    {
        int n = SEQ * HIDDEN;
        rope_pack_kernel<<<(n + 255) / 256, 256, 0, stream>>>(
            Qf, Kf, Vf, pos, Qb, Kb, VTb);
    }
    flash_attn_kernel<<<dim3(NHEADS, SEQ / 64), 128, 0, stream>>>(
        Qb, Kb, VTb, attnb);
    gemm_bf16_kernel<<<dim3(HIDDEN / 64, SEQ / 64), 128, 0, stream>>>(
        attnb, Wob, out, HIDDEN, HIDDEN);
}
